// WSDDN_41351945126034
// MI455X (gfx1250) — matrix operand order repaired and checked by reading, not compiled, not run on hardware
//
#include <hip/hip_runtime.h>
#include <hip/hip_bf16.h>
#include <math.h>

typedef __attribute__((ext_vector_type(16))) _Float16 v16h;
typedef __attribute__((ext_vector_type(8)))  _Float16 v8h;
typedef __attribute__((ext_vector_type(8)))  float    v8f;
typedef int v4i __attribute__((vector_size(16)));   // GCC-style vector: matches builtin param type

#define LA_STRIDE 40   // conv tiles: 32-deep K, 80B rows (16B aligned, conflict-free)
#define LB_STRIDE 72   // conv B tile [k][n]: 144B rows
#define FSTR      72   // fc tiles: 64 halves + 8 pad, 144B rows (16B aligned, conflict-free)
#define FK        64   // fc staged K depth

// ---------- CDNA5 async global->LDS path (guarded; falls back to direct copies) ----------
#if defined(__has_builtin)
#  if __has_builtin(__builtin_amdgcn_global_load_async_to_lds_b128)
#    define USE_ASYNC 1
#  endif
#endif
#ifndef USE_ASYNC
#  define USE_ASYNC 0
#endif

#if USE_ASYNC
// (global src, lds dst, imm offset, imm cpol)
#define ASYNC_COPY16(g, l)                                                     \
    __builtin_amdgcn_global_load_async_to_lds_b128((v4i*)(g), (v4i*)(l), 0, 0)
#else
#define ASYNC_COPY16(g, l) (*(v8h*)(l) = *(const v8h*)(g))
#endif

__device__ __forceinline__ void async_wait_all() {
#if USE_ASYNC
#  if __has_builtin(__builtin_amdgcn_s_wait_asynccnt)
    __builtin_amdgcn_s_wait_asynccnt(0);
#  else
    asm volatile("s_wait_asynccnt 0x0" ::: "memory");
#  endif
#endif
}

// ---------------- WMMA fragment loaders (LDS -> VGPR) ----------------
// A (16x32 f16) from [m][k] tile: lane L holds M = msub*16 + (L&15);
// regs 0..3 -> K=kb..kb+7, regs 4..7 -> K=16+kb..+7, kb=(L>>4)*8: two 16B LDS loads.
__device__ __forceinline__ v16h load_a_frag_s(const _Float16* la, int stride, int koff,
                                              int lane, int msub) {
    const int row = msub * 16 + (lane & 15);
    const int kb  = (lane >> 4) << 3;
    const _Float16* p = &la[row * stride + koff + kb];
    v8h lo = *(const v8h*)p;
    v8h hi = *(const v8h*)(p + 16);
    v16h a;
#pragma unroll
    for (int i = 0; i < 8; ++i) { a[i] = lo[i]; a[i + 8] = hi[i]; }
    return a;
}

// B (32x16 f16) from [k][n] tile: lane L holds N = nsub*16 + (L&15);
// regs 0..3 -> K=kb..kb+7, regs 4..7 -> K=16+kb..+7, kb=(L>>4)*8 (strided LDS reads).
__device__ __forceinline__ v16h load_b_rows(const _Float16* lb, int stride, int krow_off,
                                            int lane, int nsub) {
    const int col = nsub * 16 + (lane & 15);
    const int kb  = krow_off + ((lane >> 4) << 3);
    v16h b;
#pragma unroll
    for (int i = 0; i < 8; ++i) { b[i] = lb[(kb + i) * stride + col]; b[i + 8] = lb[(kb + 16 + i) * stride + col]; }
    return b;
}

// ---------------- FC: out_f16 = relu(X_f16 @ WT_f16 + b) ----------------
// X is [M][K]; WT is PRE-TRANSPOSED weights [K][N] so both operands stage as
// contiguous b128 async copies and A fragments are ds_load_b128 pairs (B uses strided reads).
// 64-deep K tiles, double-buffered LDS, GLOBAL_LOAD_ASYNC_TO_LDS_B128 + ASYNCcnt.
__global__ __launch_bounds__(256) void fc_wmma_f16(
    const _Float16* __restrict__ X, const _Float16* __restrict__ WT,
    const float* __restrict__ bias, _Float16* __restrict__ out,
    int N, int K)
{
    const int n0 = blockIdx.x * 64;
    const int m0 = blockIdx.y * 64;

    __shared__ __align__(16) _Float16 la[2][64 * FSTR];   // [m][k]
    __shared__ __align__(16) _Float16 lb[2][64 * FSTR];   // [k][n]

    const int tid  = threadIdx.x;
    const int lane = tid & 31;
    const int wave = tid >> 5;
    const int wm   = wave & 3;          // M subtile 0..3
    const int wn0  = (wave >> 2) << 1;  // N subtiles {0,1} or {2,3}

    v8f acc0 = {}; v8f acc1 = {};

    const int row = tid >> 2;           // 0..63 (m-row for A, k-row for B)
    const int seg = (tid & 3) << 3;     // halves 0/8/16/24; +32 for second chunk

#define FC_STAGE(bufidx, kk0)                                                  \
    do {                                                                       \
        const _Float16* gA = &X [(long)(m0 + row) * K + (kk0) + seg];          \
        const _Float16* gB = &WT[(long)((kk0) + row) * N + n0 + seg];          \
        _Float16* lA = &la[bufidx][row * FSTR + seg];                          \
        _Float16* lB = &lb[bufidx][row * FSTR + seg];                          \
        ASYNC_COPY16(gA,      lA);                                             \
        ASYNC_COPY16(gA + 32, lA + 32);                                        \
        ASYNC_COPY16(gB,      lB);                                             \
        ASYNC_COPY16(gB + 32, lB + 32);                                        \
    } while (0)

    FC_STAGE(0, 0);
    for (int k0 = 0; k0 < K; k0 += FK) {
        const int cur = (k0 >> 6) & 1;
        async_wait_all();          // this wave's copies for 'cur' have landed
        __syncthreads();           // everyone's copies visible; prev reads of cur^1 done
        if (k0 + FK < K) FC_STAGE(cur ^ 1, k0 + FK);   // overlap with WMMAs below

#pragma unroll
        for (int s = 0; s < 2; ++s) {
            const int koff = s * 32;
            const v16h af  = load_a_frag_s(la[cur], FSTR, koff, lane, wm);
            const v16h bf0 = load_b_rows(lb[cur], FSTR, koff, lane, wn0);
            const v16h bf1 = load_b_rows(lb[cur], FSTR, koff, lane, wn0 + 1);
            acc0 = __builtin_amdgcn_wmma_f32_16x16x32_f16(false, af, false, bf0, (short)0, acc0, false, false);
            acc1 = __builtin_amdgcn_wmma_f32_16x16x32_f16(false, af, false, bf1, (short)0, acc1, false, false);
        }
        __syncthreads();
    }
#undef FC_STAGE

    const int nl = lane & 15;
    const int mh = (lane >> 4) << 3;
#pragma unroll
    for (int r = 0; r < 8; ++r) {
        const int mrow = m0 + wm * 16 + mh + r;
        const int e0 = n0 + wn0 * 16 + nl;
        const int e1 = e0 + 16;
        out[(long)mrow * N + e0] = (_Float16)fmaxf(acc0[r] + bias[e0], 0.0f);
        out[(long)mrow * N + e1] = (_Float16)fmaxf(acc1[r] + bias[e1], 0.0f);
    }
}

// ---------------- Conv: implicit GEMM, M=Cout, N=H*W, K=Cin*9 (f32 in, f16 WMMA) ----------------
__global__ __launch_bounds__(256) void conv_wmma(
    const float* __restrict__ in, const float* __restrict__ wgt,
    const float* __restrict__ bias, const float* __restrict__ scl,
    const float* __restrict__ sft, float* __restrict__ out,
    int Cin, int Cout, int H, int W)
{
    const int HW   = H * W;
    const int Ktot = Cin * 9;
    const int n0   = blockIdx.x * 64;
    const int m0   = blockIdx.y * 64;

    __shared__ __align__(16) _Float16 la[64 * LA_STRIDE];
    __shared__ __align__(16) _Float16 lb[32 * LB_STRIDE];

    const int tid  = threadIdx.x;
    const int lane = tid & 31;
    const int wave = tid >> 5;
    const int wm   = wave & 3;
    const int wn0  = (wave >> 2) << 1;

    v8f acc0 = {}; v8f acc1 = {};

    const int arow = tid >> 2;
    const int akk  = (tid & 3) << 3;
    const int bcol = tid & 63;
    const int bkk  = (tid >> 6) << 3;
    const int pix  = n0 + bcol;
    const int oy   = (pix < HW) ? (pix / W) : 0;
    const int ox   = (pix < HW) ? (pix - oy * W) : 0;

    for (int k0 = 0; k0 < Ktot; k0 += 32) {
#pragma unroll
        for (int i = 0; i < 8; ++i) {
            const int kg = k0 + akk + i;
            const float v = (kg < Ktot) ? wgt[(m0 + arow) * Ktot + kg] : 0.0f;
            la[arow * LA_STRIDE + akk + i] = (_Float16)v;
        }
#pragma unroll
        for (int i = 0; i < 8; ++i) {
            const int kg = k0 + bkk + i;
            float v = 0.0f;
            if (kg < Ktot && pix < HW) {
                const int cin = kg / 9;
                const int rem = kg - cin * 9;
                const int r   = rem / 3;
                const int s   = rem - r * 3;
                const int iy  = oy + r - 1;
                const int ix  = ox + s - 1;
                if (iy >= 0 && iy < H && ix >= 0 && ix < W)
                    v = in[cin * HW + iy * W + ix];
            }
            lb[(bkk + i) * LB_STRIDE + bcol] = (_Float16)v;
        }
        __syncthreads();

        const v16h af  = load_a_frag_s(la, LA_STRIDE, 0, lane, wm);
        const v16h bf0 = load_b_rows(lb, LB_STRIDE, 0, lane, wn0);
        const v16h bf1 = load_b_rows(lb, LB_STRIDE, 0, lane, wn0 + 1);
        acc0 = __builtin_amdgcn_wmma_f32_16x16x32_f16(false, af, false, bf0, (short)0, acc0, false, false);
        acc1 = __builtin_amdgcn_wmma_f32_16x16x32_f16(false, af, false, bf1, (short)0, acc1, false, false);
        __syncthreads();
    }

    const int nl = lane & 15;
    const int mh = (lane >> 4) << 3;
#pragma unroll
    for (int r = 0; r < 8; ++r) {
        const int coutg = m0 + wm * 16 + mh + r;
        const float bb = bias[coutg], ss = scl[coutg], tt = sft[coutg];
        const int p0 = n0 + wn0 * 16 + nl;
        const int p1 = p0 + 16;
        if (p0 < HW) out[coutg * HW + p0] = fmaxf((acc0[r] + bb) * ss + tt, 0.0f);
        if (p1 < HW) out[coutg * HW + p1] = fmaxf((acc1[r] + bb) * ss + tt, 0.0f);
    }
}

// ---------------- f32 [N][K] -> f16 [K][N] tiled transpose (fc weights) ----------------
__global__ __launch_bounds__(256) void cvt_transpose_f16(
    const float* __restrict__ in, _Float16* __restrict__ out, int N, int K)
{
    __shared__ float tile[32][33];
    const int tx = threadIdx.x & 31;
    const int ty = threadIdx.x >> 5;        // 0..7
    const int n0 = blockIdx.x * 32;
    const int k0 = blockIdx.y * 32;
#pragma unroll
    for (int j = 0; j < 4; ++j)
        tile[ty + 8 * j][tx] = in[(long)(n0 + ty + 8 * j) * K + k0 + tx];
    __syncthreads();
#pragma unroll
    for (int j = 0; j < 4; ++j)
        out[(long)(k0 + ty + 8 * j) * N + n0 + tx] = (_Float16)tile[tx][ty + 8 * j];
}

// ---------------- 2x2 max pool ----------------
__global__ void maxpool2(const float* __restrict__ in, float* __restrict__ out,
                         int C, int Ho, int Wo)
{
    const int i = blockIdx.x * blockDim.x + threadIdx.x;
    const int tot = C * Ho * Wo;
    if (i >= tot) return;
    const int wo = i % Wo;
    const int t  = i / Wo;
    const int ho = t % Ho;
    const int c  = t / Ho;
    const int Wi = Wo * 2;
    const float* p = in + (size_t)c * (Ho * 2) * Wi + (ho * 2) * Wi + wo * 2;
    out[i] = fmaxf(fmaxf(p[0], p[1]), fmaxf(p[Wi], p[Wi + 1]));
}

// ---------------- ROI SPP -> duplicated 4096-wide f16 features ----------------
__global__ void roi_spp(const float* __restrict__ fm, const int* __restrict__ ssw,
                        _Float16* __restrict__ yout)
{
    const int i = blockIdx.x * blockDim.x + threadIdx.x;
    if (i >= 512 * 512) return;
    const int roi = i >> 9;
    const int c   = i & 511;

    const int y0 = ssw[roi * 4 + 0];
    const int x0 = ssw[roi * 4 + 1];
    const int hh = ssw[roi * 4 + 2];
    const int ww = ssw[roi * 4 + 3];

    const int widh = (hh + 1) >> 1;
    const int padh = min((widh * 2 - hh + 1) >> 1, widh >> 1);
    const int widw = (ww + 1) >> 1;
    const int padw = min((widw * 2 - ww + 1) >> 1, widw >> 1);

    int rs[2], re[2], cs[2], ce[2];
#pragma unroll
    for (int a = 0; a < 2; ++a) {
        int st = y0 + a * widh - padh;          // unclamped start
        re[a] = min(st + widh, y0 + hh);        // stop uses unclamped start
        rs[a] = max(st, y0);
        st = x0 + a * widw - padw;
        ce[a] = min(st + widw, x0 + ww);
        cs[a] = max(st, x0);
    }

    const float* f = fm + c * 196;  // 14x14
#pragma unroll
    for (int a = 0; a < 2; ++a) {
#pragma unroll
        for (int b = 0; b < 2; ++b) {
            float m = -INFINITY;
            for (int iy = rs[a]; iy < re[a]; ++iy)
                for (int ix = cs[b]; ix < ce[b]; ++ix)
                    m = fmaxf(m, f[iy * 14 + ix]);
            const int o = roi * 4096 + c * 4 + a * 2 + b;
            const _Float16 mh = (_Float16)m;
            yout[o] = mh;
            yout[o + 2048] = mh;
        }
    }
}

// ---------------- fc8 heads: 40 length-4096 dots per ROI (wave reductions) ----------------
__global__ __launch_bounds__(256) void fc8_heads(
    const _Float16* __restrict__ h,
    const float* __restrict__ wc, const float* __restrict__ bc,
    const float* __restrict__ wd, const float* __restrict__ bd,
    float* __restrict__ xc, float* __restrict__ xd)
{
    const int r    = blockIdx.x;
    const int wave = threadIdx.x >> 5;
    const int lane = threadIdx.x & 31;
    const _Float16* hr = h + (size_t)r * 4096;

    for (int j = wave; j < 40; j += 8) {
        const float* wrow = (j < 20) ? (wc + (size_t)j * 4096) : (wd + (size_t)(j - 20) * 4096);
        float s = 0.0f;
        for (int d = lane; d < 4096; d += 32) s += (float)hr[d] * wrow[d];
#pragma unroll
        for (int off = 16; off > 0; off >>= 1) s += __shfl_xor(s, off, 32);
        if (lane == 0) {
            if (j < 20) xc[r * 20 + j]        = fmaxf(s + bc[j], 0.0f);
            else        xd[r * 20 + (j - 20)] = fmaxf(s + bd[j - 20], 0.0f);
        }
    }
}

// softmax over classes (per roi)
__global__ void softmax_clf(const float* __restrict__ xc, float* __restrict__ sc)
{
    const int r = blockIdx.x * blockDim.x + threadIdx.x;
    if (r >= 512) return;
    float m = -INFINITY;
#pragma unroll
    for (int c = 0; c < 20; ++c) m = fmaxf(m, xc[r * 20 + c]);
    float e[20], s = 0.0f;
#pragma unroll
    for (int c = 0; c < 20; ++c) { e[c] = __expf(xc[r * 20 + c] - m); s += e[c]; }
#pragma unroll
    for (int c = 0; c < 20; ++c) sc[r * 20 + c] = e[c] / s;
}

// softmax over ROIs (per class)
__global__ __launch_bounds__(256) void softmax_dct(const float* __restrict__ xd, float* __restrict__ sd)
{
    const int c = blockIdx.x;
    const int t = threadIdx.x;
    __shared__ float red[256];

    float m = -INFINITY;
    for (int r = t; r < 512; r += 256) m = fmaxf(m, xd[r * 20 + c]);
    red[t] = m; __syncthreads();
    for (int s = 128; s > 0; s >>= 1) { if (t < s) red[t] = fmaxf(red[t], red[t + s]); __syncthreads(); }
    m = red[0]; __syncthreads();

    float sum = 0.0f;
    for (int r = t; r < 512; r += 256) sum += __expf(xd[r * 20 + c] - m);
    red[t] = sum; __syncthreads();
    for (int s = 128; s > 0; s >>= 1) { if (t < s) red[t] += red[t + s]; __syncthreads(); }
    sum = red[0];

    for (int r = t; r < 512; r += 256) sd[r * 20 + c] = __expf(xd[r * 20 + c] - m) / sum;
}

// scores[c] = sum_r sigma_clf[r,c] * sigma_dct[r,c]
__global__ __launch_bounds__(256) void score_sum(
    const float* __restrict__ sc, const float* __restrict__ sd, float* __restrict__ out)
{
    const int c = blockIdx.x;
    const int t = threadIdx.x;
    __shared__ float red[256];
    float s = 0.0f;
    for (int r = t; r < 512; r += 256) s += sc[r * 20 + c] * sd[r * 20 + c];
    red[t] = s; __syncthreads();
    for (int k = 128; k > 0; k >>= 1) { if (t < k) red[t] += red[t + k]; __syncthreads(); }
    if (t == 0) out[c] = red[0];
}

// ---------------- launch ----------------
extern "C" void kernel_launch(void* const* d_in, const int* in_sizes, int n_in,
                              void* d_out, int out_size, void* d_ws, size_t ws_size,
                              hipStream_t stream)
{
    const float* x   = (const float*)d_in[0];
    const int*   ssw = (const int*)d_in[1];
    // conv_params: d_in[2 + 4*L + {w,b,s,t}], fc_params at d_in[34..41]
    const float* fc6w = (const float*)d_in[34]; const float* fc6b = (const float*)d_in[35];
    const float* fc7w = (const float*)d_in[36]; const float* fc7b = (const float*)d_in[37];
    const float* fc8cw = (const float*)d_in[38]; const float* fc8cb = (const float*)d_in[39];
    const float* fc8dw = (const float*)d_in[40]; const float* fc8db = (const float*)d_in[41];

    char* base = (char*)d_ws;
    const size_t BUFB = 64ull * 224 * 224 * 4;           // 12,845,056 B per conv buffer
    float*    buf0 = (float*)base;
    float*    buf1 = (float*)(base + BUFB);
    _Float16* yh   = (_Float16*)(base + 2 * BUFB);                       // 512x4096 f16
    _Float16* whT  = (_Float16*)(base + 2 * BUFB + 4194304);             // 4096x4096 f16 transposed (fc6 then fc7)
    _Float16* h1h  = (_Float16*)(base + 2 * BUFB + 4194304 + 33554432);
    _Float16* h2h  = (_Float16*)(base + 2 * BUFB + 2 * 4194304 + 33554432);
    float*    xc   = (float*)(base + 2 * BUFB + 3 * 4194304 + 33554432);
    float*    xd   = xc + 512 * 20;

    auto conv = [&](const float* in, int L, float* out, int Cin, int Cout, int H, int W) {
        dim3 g((H * W + 63) / 64, Cout / 64);
        conv_wmma<<<g, 256, 0, stream>>>(in, (const float*)d_in[2 + 4 * L],
                                         (const float*)d_in[3 + 4 * L],
                                         (const float*)d_in[4 + 4 * L],
                                         (const float*)d_in[5 + 4 * L],
                                         out, Cin, Cout, H, W);
    };
    auto pool = [&](const float* in, float* out, int C, int Ho, int Wo) {
        int tot = C * Ho * Wo;
        maxpool2<<<(tot + 255) / 256, 256, 0, stream>>>(in, out, C, Ho, Wo);
    };

    conv(x,    0, buf0,   3,  64, 224, 224);  pool(buf0, buf1,  64, 112, 112);
    conv(buf1, 1, buf0,  64, 128, 112, 112);  pool(buf0, buf1, 128,  56,  56);
    conv(buf1, 2, buf0, 128, 256,  56,  56);
    conv(buf0, 3, buf1, 256, 256,  56,  56);  pool(buf1, buf0, 256,  28,  28);
    conv(buf0, 4, buf1, 256, 512,  28,  28);
    conv(buf1, 5, buf0, 512, 512,  28,  28);  pool(buf0, buf1, 512,  14,  14);
    conv(buf1, 6, buf0, 512, 512,  14,  14);
    conv(buf0, 7, buf1, 512, 512,  14,  14);  // final feature map 512x14x14

    roi_spp<<<(512 * 512 + 255) / 256, 256, 0, stream>>>(buf1, ssw, yh);

    const dim3 tg(4096 / 32, 4096 / 32);
    cvt_transpose_f16<<<tg, 256, 0, stream>>>(fc6w, whT, 4096, 4096);
    fc_wmma_f16<<<dim3(4096 / 64, 512 / 64), 256, 0, stream>>>(yh,  whT, fc6b, h1h, 4096, 4096);
    cvt_transpose_f16<<<tg, 256, 0, stream>>>(fc7w, whT, 4096, 4096);
    fc_wmma_f16<<<dim3(4096 / 64, 512 / 64), 256, 0, stream>>>(h1h, whT, fc7b, h2h, 4096, 4096);

    fc8_heads<<<512, 256, 0, stream>>>(h2h, fc8cw, fc8cb, fc8dw, fc8db, xc, xd);

    float* out    = (float*)d_out;
    float* sc_out = out + 20;            // sigma_clf (512x20)
    float* sd_out = out + 20 + 10240;    // sigma_dct (512x20)

    softmax_clf<<<2, 256, 0, stream>>>(xc, sc_out);
    softmax_dct<<<20, 256, 0, stream>>>(xd, sd_out);
    score_sum<<<20, 256, 0, stream>>>(sc_out, sd_out, out);
}
